// PeriodicPrimitives2D_82978768159508
// MI455X (gfx1250) — matrix operand order repaired and checked by reading, not compiled, not run on hardware
//
#include <hip/hip_runtime.h>
#include <stdint.h>

// Problem constants (match reference)
#define NPIX   262144
#define NG     4096
#define NK     4
#define NCH    3
#define TG     128        // gaussians staged per LDS tile
#define PSTR   24         // packed floats per gaussian
#define BCHUNK 512        // halves per 32-gaussian B-operand chunk (32 lanes x 16)
#define BLOCK  256        // 8 waves; each wave owns 16 pixels

typedef __attribute__((ext_vector_type(16))) _Float16 v16h;
typedef __attribute__((ext_vector_type(8)))  float    v8f;
typedef __attribute__((ext_vector_type(4)))  float    v4f;
typedef int b128_t __attribute__((vector_size(16)));   // matches builtin's V4i pointee

#if defined(__gfx1250__) && __has_builtin(__builtin_amdgcn_global_load_async_to_lds_b128)
#define USE_ASYNC_LDS 1
#else
#define USE_ASYNC_LDS 0
#endif

// v_cos_f32 has period 1.0 (cos(2*pi*x)) -> pass t*f directly, no 2*pi multiply.
__device__ __forceinline__ float cos_turns(float t) { return __builtin_amdgcn_cosf(t); }
// v_exp_f32 is exp2; exp(-0.5*s) == exp2(-0.72134752*s)
__device__ __forceinline__ float gauss_exp(float s) { return __builtin_amdgcn_exp2f(-0.72134752f * s); }

#if USE_ASYNC_LDS
__device__ __forceinline__ void async16(const void* g, void* l) {
  // generic->AS1: value-preserving; generic->AS3: low 32 bits are the LDS byte offset
  __builtin_amdgcn_global_load_async_to_lds_b128(
      (__attribute__((address_space(1))) b128_t*)(unsigned long long)(uintptr_t)g,
      (__attribute__((address_space(3))) b128_t*)(unsigned)(uintptr_t)l,
      0, 0);
}
__device__ __forceinline__ void wait_async0() {
#if __has_builtin(__builtin_amdgcn_s_wait_asynccnt)
  __builtin_amdgcn_s_wait_asynccnt(0);
#else
  asm volatile("s_wait_asynccnt 0" ::: "memory");
#endif
}
#endif

// ---------------- prep: pack per-gaussian invariants ----------------
// params, 24 floats per gaussian:
//  [0]=px [1]=py [2]=cos(rot) [3]=sin(rot) [4]=exp(sx) [5]=exp(sy) [6,7]=pad
//  [8..11]=fx_k (idx*MAXF/NF)  [12..15]=fy_k  [16..19]=cx_k  [20..23]=cy_k
// B-operand image, per 32-gaussian chunk: pb[chunk][lane][i] = f16 B fragment
//  element i of lane's v16h = color[K = (lane>>4)*16 + i][N = lane&15], 0 for N>=3
__global__ void prep_kernel(const float* __restrict__ col,
                            const float* __restrict__ pos,
                            const float* __restrict__ scl,
                            const float* __restrict__ rot,
                            const float* __restrict__ cf,
                            const int*   __restrict__ idx,
                            float* __restrict__ pp,
                            _Float16* __restrict__ pb)
{
  int g = blockIdx.x * blockDim.x + threadIdx.x;
  if (g >= NG) return;
  float r  = rot[g];
  float cr = __cosf(r), sr = __sinf(r);
  float* o = pp + g * PSTR;
  o[0] = pos[2*g+0]; o[1] = pos[2*g+1];
  o[2] = cr;         o[3] = sr;
  o[4] = __expf(scl[2*g+0]);
  o[5] = __expf(scl[2*g+1]);
  o[6] = 0.0f;       o[7] = 0.0f;
  const float fscale = 1024.0f / 1024.0f;   // MAXF / NF
  #pragma unroll
  for (int k = 0; k < NK; ++k) {
    o[8  + k] = (float)idx[(g*NK + k)*2 + 0] * fscale;
    o[12 + k] = (float)idx[(g*NK + k)*2 + 1] * fscale;
    o[16 + k] = cf[(g*NK + k)*2 + 0];
    o[20 + k] = cf[(g*NK + k)*2 + 1];
  }
  // scatter this gaussian's colors into the B-operand image of its chunk
  _Float16* ob = pb + (size_t)(g >> 5) * BCHUNK;
  int laneBase = g & 16;        // K=g%32: rows 0..15 -> lanes 0..15, rows 16..31 -> lanes 16..31
  int i = g & 15;               // element index within the lane's v16h
  #pragma unroll
  for (int l = 0; l < 16; ++l)
    ob[(laneBase + l) * 16 + i] = (l < NCH) ? (_Float16)col[g*3 + l] : (_Float16)0.0f;
}

// ---------------- render: wave = 16 pixels, WMMA over gaussian chunks ----------------
__global__ __launch_bounds__(BLOCK)
void render_kernel(const float* __restrict__ x,
                   const float* __restrict__ pp,
                   const _Float16* __restrict__ pb,
                   float* __restrict__ out)
{
  __shared__ __attribute__((aligned(32))) float    sP[2][TG * PSTR];
  __shared__ __attribute__((aligned(32))) _Float16 sB[2][(TG / 32) * BCHUNK];

  const int tid  = threadIdx.x;
  const int wave = tid >> 5;
  const int lane = tid & 31;
  const int half = lane >> 4;      // which half-wave (K split)
  const int mrow = lane & 15;      // A-matrix row (pixel) for this lane
  const int chan = lane & 15;      // C column (channel) for this lane

  const int pixBase = blockIdx.x * (BLOCK / 32 * 16) + wave * 16;
  const float px = x[2 * (pixBase + mrow) + 0];
  const float py = x[2 * (pixBase + mrow) + 1];

  v8f acc = {};

  // stage one tile (params + B image) into LDS buffer `buf`
  auto stage = [&](int buf, int gt) {
    const float*    gp = pp + (size_t)gt * PSTR;
    const _Float16* gb = pb + (size_t)(gt >> 5) * BCHUNK;
#if USE_ASYNC_LDS
    for (int c = tid; c < TG * PSTR / 4; c += BLOCK)
      async16(gp + c * 4, &sP[buf][c * 4]);
    for (int c = tid; c < (TG / 32) * BCHUNK / 8; c += BLOCK)
      async16(gb + c * 8, &sB[buf][c * 8]);
#else
    for (int c = tid; c < TG * PSTR / 4; c += BLOCK)
      ((v4f*)sP[buf])[c] = ((const v4f*)gp)[c];
    for (int c = tid; c < (TG / 32) * BCHUNK / 8; c += BLOCK)
      ((v4f*)sB[buf])[c] = ((const v4f*)gb)[c];
#endif
  };

  const int ntiles = NG / TG;
  stage(0, 0);
#if USE_ASYNC_LDS
  wait_async0();
#endif
  __syncthreads();

  #pragma unroll 1
  for (int t = 0; t < ntiles; ++t) {
    const int buf = t & 1;
    if (t + 1 < ntiles) stage(buf ^ 1, (t + 1) * TG);   // overlap DMA with compute

    #pragma unroll 1
    for (int sub = 0; sub < TG; sub += 32) {
      // B fragment: one aligned 32B LDS read (2x ds_load_b128), zero divergence
      const v16h b = *(const v16h*)&sB[buf][((sub >> 5) * BCHUNK) + (lane << 4)];

      // A matrix (16x32 f16): lane holds row M=mrow; element j covers
      // K = j + 16*half  (contiguous order, matches B image from prep)
      v16h a;
      #pragma unroll
      for (int j = 0; j < 16; ++j) {
        int k = j + half * 16;
        const v4f* q = (const v4f*)&sP[buf][(sub + k) * PSTR];
        v4f q0  = q[0];            // px py cos sin
        v4f q1  = q[1];            // inv_sx inv_sy pad pad
        v4f qfx = q[2];
        v4f qfy = q[3];
        v4f qcx = q[4];
        v4f qcy = q[5];

        float dx = px - q0.x;
        float dy = py - q0.y;
        float tx =  dx * q0.z + dy * q0.w;
        float ty = -dx * q0.w + dy * q0.z;
        float ax = tx * q1.x, ay = ty * q1.y;
        float e  = gauss_exp(ax * ax + ay * ay);

        float wx = qcx.x * cos_turns(tx * qfx.x)
                 + qcx.y * cos_turns(tx * qfx.y)
                 + qcx.z * cos_turns(tx * qfx.z)
                 + qcx.w * cos_turns(tx * qfx.w);
        float wy = qcy.x * cos_turns(ty * qfy.x)
                 + qcy.y * cos_turns(ty * qfy.y)
                 + qcy.z * cos_turns(ty * qfy.z)
                 + qcy.w * cos_turns(ty * qfy.w);

        a[j] = (_Float16)(e * wx * wy);
      }

      acc = __builtin_amdgcn_wmma_f32_16x16x32_f16(
          /*neg_a=*/false, a, /*neg_b=*/false, b,
          /*c_mod=*/(short)0, acc, /*reuse_a=*/false, /*reuse_b=*/false);
    }

#if USE_ASYNC_LDS
    wait_async0();     // next tile fully landed in LDS
#endif
    __syncthreads();   // all waves done with current tile before it is overwritten
  }

  // C/D layout: VGPR r -> M = r + 8*half, N = chan
  if (chan < NCH) {
    #pragma unroll
    for (int r = 0; r < 8; ++r) {
      int pix = pixBase + half * 8 + r;
      out[pix * NCH + chan] = acc[r];
    }
  }
}

extern "C" void kernel_launch(void* const* d_in, const int* in_sizes, int n_in,
                              void* d_out, int out_size, void* d_ws, size_t ws_size,
                              hipStream_t stream) {
  const float* x   = (const float*)d_in[0];
  const float* col = (const float*)d_in[1];
  const float* pos = (const float*)d_in[2];
  const float* scl = (const float*)d_in[3];
  const float* rot = (const float*)d_in[4];
  const float* cf  = (const float*)d_in[5];
  const int*   idx = (const int*)d_in[6];

  float*    pp = (float*)d_ws;
  _Float16* pb = (_Float16*)((char*)d_ws + (size_t)NG * PSTR * sizeof(float));

  prep_kernel<<<NG / 256, 256, 0, stream>>>(col, pos, scl, rot, cf, idx, pp, pb);
  render_kernel<<<NPIX / (BLOCK / 32 * 16), BLOCK, 0, stream>>>(x, pp, pb, (float*)d_out);
}
